// GAT_16097537425901
// MI455X (gfx1250) — matrix operand order repaired and checked by reading, not compiled, not run on hardware
//
#include <hip/hip_runtime.h>
#include <math.h>

// ---------------------------------------------------------------------------
// CDNA5 (gfx1250) GAT forward: wave32, WMMA f32_16x16x32_f16 for both GEMMs,
// exact-f32 edge softmax/aggregation with global float atomics.
// ---------------------------------------------------------------------------

typedef __attribute__((ext_vector_type(16))) _Float16 v16h;
typedef __attribute__((ext_vector_type(8)))  _Float16 v8h;
typedef __attribute__((ext_vector_type(8)))  float    v8f;

#define NEG_SLOPE 0.2f
#define SM_EPS    1e-16f

// ---------------- float atomic max via int/uint bit trick -------------------
__device__ __forceinline__ void atomicMaxF(float* addr, float val) {
    if (val >= 0.0f) {
        atomicMax((int*)addr, __float_as_int(val));
    } else {
        atomicMin((unsigned int*)addr, (unsigned int)__float_as_int(val));
    }
}

// ---------------- small utility kernels -------------------------------------
__global__ void fill_kernel(float* __restrict__ p, float v, long long n) {
    long long i = (long long)blockIdx.x * blockDim.x + threadIdx.x;
    if (i < n) p[i] = v;
}

__global__ void cvt_f16_kernel(const float* __restrict__ in, _Float16* __restrict__ out,
                               long long n) {
    long long i = (long long)blockIdx.x * blockDim.x + threadIdx.x;
    if (i < n) out[i] = (_Float16)in[i];
}

// W [K, Nw] row-major f32  ->  Wt [Nw, K] f16 (each output column contiguous)
__global__ void cvt_w_t_kernel(const float* __restrict__ W, _Float16* __restrict__ Wt,
                               int K, int Nw) {
    int idx = blockIdx.x * blockDim.x + threadIdx.x;
    if (idx >= K * Nw) return;
    int k = idx / Nw, n = idx % Nw;
    Wt[(size_t)n * K + k] = (_Float16)W[idx];
}

// ---------------- WMMA GEMM: C[M,Nw] = A[M,K] (f16) * Bt[Nw,K]^T (f16) ------
// One wave per 16x16 output tile; K in steps of 32 via v_wmma_f32_16x16x32_f16.
__global__ __launch_bounds__(128)
void wmma_gemm_kernel(const _Float16* __restrict__ A,   // [M,K] row-major
                      const _Float16* __restrict__ Bt,  // [Nw,K] (col-major W)
                      float* __restrict__ C,            // [M,Nw] row-major
                      int M, int Nw, int K) {
    const int lane   = threadIdx.x & 31;
    const int wave   = threadIdx.x >> 5;
    const int tilesN = Nw >> 4;
    const int Mt     = (M + 15) >> 4;
    const int wid    = blockIdx.x * 4 + wave;
    if (wid >= Mt * tilesN) return;
    const int tM  = wid / tilesN;
    const int tN  = wid % tilesN;
    const int sel = lane >> 4;      // 0: lanes 0-15, 1: lanes 16-31
    const int l15 = lane & 15;

    int rowA = tM * 16 + l15;
    if (rowA >= M) rowA = M - 1;    // tail-safe clamp (masked at store)
    const _Float16* Abase = A  + (size_t)rowA * K;
    const _Float16* Bbase = Bt + (size_t)(tN * 16 + l15) * K;

    v8f c = {};
    for (int k0 = 0; k0 < K; k0 += 32) {
        // A fragment (16x32 f16): lanes 0-15 hold K = k0..k0+7 & k0+16..k0+23,
        // lanes 16-31 hold K = k0+8..k0+15 & k0+24..k0+31 (ISA 7.12.2).
        v16h a, b;
        ((v8h*)&a)[0] = *(const v8h*)(Abase + k0 + sel * 8);
        ((v8h*)&a)[1] = *(const v8h*)(Abase + k0 + 16 + sel * 8);
        // B fragment (32x16 f16): lanes 0-15 hold K = k0..k0+7 & k0+16..k0+23,
        // lanes 16-31 hold K = k0+8..k0+15 & k0+24..k0+31, column n contiguous in transposed weight.
        ((v8h*)&b)[0] = *(const v8h*)(Bbase + k0 + sel * 8);
        ((v8h*)&b)[1] = *(const v8h*)(Bbase + k0 + 16 + sel * 8);
        if (k0 + 32 < K) __builtin_prefetch(Abase + k0 + 32, 0, 0);
        c = __builtin_amdgcn_wmma_f32_16x16x32_f16(
                false, a, false, b, (short)0, c, false, false);
    }
    // C layout: VGPR v -> row v + sel*8, col = lane&15.
    float* Cb = C + (size_t)(tM * 16 + sel * 8) * Nw + tN * 16 + l15;
#pragma unroll
    for (int v = 0; v < 8; ++v) {
        int r = tM * 16 + sel * 8 + v;
        if (r < M) Cb[(size_t)v * Nw] = c[v];
    }
}

// ---------------- per-node attention logits ---------------------------------
// aS[n,h] = sum_d feat[n,h*D+d]*attS[h,d]; same for aD.
__global__ void alpha_kernel(const float* __restrict__ feat,
                             const float* __restrict__ attS,
                             const float* __restrict__ attD,
                             float* __restrict__ aS, float* __restrict__ aD,
                             long long N, int H, int D) {
    long long idx = (long long)blockIdx.x * blockDim.x + threadIdx.x;
    if (idx >= N * H) return;
    long long n = idx / H;
    int h = (int)(idx % H);
    const float* f = feat + (size_t)n * H * D + (size_t)h * D;
    const float* s = attS + (size_t)h * D;
    const float* d = attD + (size_t)h * D;
    float ss = 0.f, sd = 0.f;
    for (int i = 0; i < D; ++i) { float v = f[i]; ss += v * s[i]; sd += v * d[i]; }
    aS[idx] = ss;
    aD[idx] = sd;
}

// ---------------- edge pass 1: segment max ----------------------------------
__global__ void edge_max_kernel(const long long* __restrict__ src,
                                const long long* __restrict__ dst,
                                long long E, long long N, int H,
                                const float* __restrict__ aS,
                                const float* __restrict__ aD,
                                float* __restrict__ m) {
    long long idx = (long long)blockIdx.x * blockDim.x + threadIdx.x;
    long long tot = (E + N) * H;
    if (idx >= tot) return;
    long long e = idx / H;
    int h = (int)(idx % H);
    long long s, d;
    if (e < E) { s = src[e]; d = dst[e]; } else { s = d = e - E; }
    float v = aS[(size_t)s * H + h] + aD[(size_t)d * H + h];
    v = v > 0.f ? v : NEG_SLOPE * v;
    atomicMaxF(&m[(size_t)d * H + h], v);
}

// ---------------- edge pass 2: exp & segment sum ----------------------------
__global__ void edge_expsum_kernel(const long long* __restrict__ src,
                                   const long long* __restrict__ dst,
                                   long long E, long long N, int H,
                                   const float* __restrict__ aS,
                                   const float* __restrict__ aD,
                                   const float* __restrict__ m,
                                   float* __restrict__ wbuf,
                                   float* __restrict__ denom) {
    long long idx = (long long)blockIdx.x * blockDim.x + threadIdx.x;
    long long tot = (E + N) * H;
    if (idx >= tot) return;
    long long e = idx / H;
    int h = (int)(idx % H);
    long long s, d;
    if (e < E) { s = src[e]; d = dst[e]; } else { s = d = e - E; }
    float v = aS[(size_t)s * H + h] + aD[(size_t)d * H + h];
    v = v > 0.f ? v : NEG_SLOPE * v;
    float w = expf(v - m[(size_t)d * H + h]);
    wbuf[idx] = w;
    atomicAdd(&denom[(size_t)d * H + h], w);
}

// ---------------- edge pass 3: weighted scatter-add -------------------------
// One thread per (edge, channel): coalesced gather of feat[src], coalesced
// float atomics into agg[dst] (both L2-resident: 51 MB each << 192 MB L2).
__global__ void edge_agg_kernel(const long long* __restrict__ src,
                                const long long* __restrict__ dst,
                                long long E, long long N, int H, int D,
                                const float* __restrict__ wbuf,
                                const float* __restrict__ denom,
                                const float* __restrict__ feat,
                                float* __restrict__ agg) {
    const int C = H * D;
    long long idx = (long long)blockIdx.x * blockDim.x + threadIdx.x;
    long long tot = (E + N) * C;
    if (idx >= tot) return;
    long long e = idx / C;
    int c = (int)(idx % C);
    int h = c / D;
    long long s, d;
    if (e < E) { s = src[e]; d = dst[e]; } else { s = d = e - E; }
    float a = wbuf[e * H + h] / (denom[(size_t)d * H + h] + SM_EPS);
    atomicAdd(&agg[(size_t)d * C + c], feat[(size_t)s * C + c] * a);
}

// ---------------- bias + ELU + f16 convert (layer-0 epilogue) ---------------
__global__ void bias_elu_cvt_kernel(const float* __restrict__ agg,
                                    const float* __restrict__ bias,
                                    _Float16* __restrict__ out,
                                    long long N, int C) {
    long long idx = (long long)blockIdx.x * blockDim.x + threadIdx.x;
    if (idx >= N * C) return;
    int c = (int)(idx % C);
    float v = agg[idx] + bias[c];
    v = v > 0.f ? v : (expf(v) - 1.f);
    out[idx] = (_Float16)v;
}

// ---------------- final bias add --------------------------------------------
__global__ void add_bias_kernel(const float* __restrict__ agg,
                                const float* __restrict__ bias,
                                float* __restrict__ out,
                                long long N, int C) {
    long long idx = (long long)blockIdx.x * blockDim.x + threadIdx.x;
    if (idx >= N * C) return;
    int c = (int)(idx % C);
    out[idx] = agg[idx] + bias[c];
}

// ---------------------------------------------------------------------------
static inline int gridFor(long long n, int b) { return (int)((n + b - 1) / b); }

extern "C" void kernel_launch(void* const* d_in, const int* in_sizes, int n_in,
                              void* d_out, int out_size, void* d_ws, size_t ws_size,
                              hipStream_t stream) {
    const float*     x     = (const float*)d_in[0];
    const long long* edge  = (const long long*)d_in[1];
    const float*     W0    = (const float*)d_in[2];
    const float*     attS0 = (const float*)d_in[3];
    const float*     attD0 = (const float*)d_in[4];
    const float*     b0    = (const float*)d_in[5];
    const float*     W1    = (const float*)d_in[6];
    const float*     attS1 = (const float*)d_in[7];
    const float*     attD1 = (const float*)d_in[8];
    const float*     b1    = (const float*)d_in[9];
    float*           out   = (float*)d_out;

    const int FIN = 256, HID = 256, H0 = 4, D0 = 64, C1 = 64;
    const long long N  = in_sizes[0] / FIN;
    const long long E  = in_sizes[1] / 2;
    const long long E2 = E + N;
    const long long* src = edge;
    const long long* dst = edge + E;

    // ---------------- workspace layout (layer-1 aliases dead layer-0 bufs) --
    char* ws = (char*)d_ws;
    size_t off = 0;
    auto alloc = [&](size_t bytes) {
        size_t o = off; off = (off + bytes + 255) & ~(size_t)255; return o;
    };
    size_t off_xh    = alloc((size_t)N * FIN * 2);   // f16 x; reused as f16 hh
    size_t off_w0t   = alloc((size_t)FIN * HID * 2); // f16 W0^T
    size_t off_w1t   = alloc((size_t)HID * C1 * 2);  // f16 W1^T
    size_t off_feat0 = alloc((size_t)N * HID * 4);   // h0 = x@W0 (region reused)
    size_t off_as0   = alloc((size_t)N * H0 * 4);
    size_t off_ad0   = alloc((size_t)N * H0 * 4);
    size_t off_m0    = alloc((size_t)N * H0 * 4);
    size_t off_den0  = alloc((size_t)N * H0 * 4);
    size_t off_wbuf  = alloc((size_t)E2 * H0 * 4);   // reused for layer 1 (H=1)
    size_t off_agg0  = alloc((size_t)N * HID * 4);
    // layer-1 buffers overlaid on feat0 region (26.4 MB <= 51.2 MB):
    size_t o1        = off_feat0;
    auto alloc1 = [&](size_t bytes) {
        size_t o = o1; o1 = (o1 + bytes + 255) & ~(size_t)255; return o;
    };
    size_t off_feat1 = alloc1((size_t)N * C1 * 4);
    size_t off_as1   = alloc1((size_t)N * 4);
    size_t off_ad1   = alloc1((size_t)N * 4);
    size_t off_m1    = alloc1((size_t)N * 4);
    size_t off_den1  = alloc1((size_t)N * 4);
    size_t off_agg1  = alloc1((size_t)N * C1 * 4);

    _Float16* xh    = (_Float16*)(ws + off_xh);
    _Float16* hh    = (_Float16*)(ws + off_xh);   // alias (x dead after GEMM0)
    _Float16* W0t   = (_Float16*)(ws + off_w0t);
    _Float16* W1t   = (_Float16*)(ws + off_w1t);
    float* feat0 = (float*)(ws + off_feat0);
    float* as0   = (float*)(ws + off_as0);
    float* ad0   = (float*)(ws + off_ad0);
    float* m0    = (float*)(ws + off_m0);
    float* den0  = (float*)(ws + off_den0);
    float* wbuf  = (float*)(ws + off_wbuf);
    float* agg0  = (float*)(ws + off_agg0);
    float* feat1 = (float*)(ws + off_feat1);
    float* as1   = (float*)(ws + off_as1);
    float* ad1   = (float*)(ws + off_ad1);
    float* m1    = (float*)(ws + off_m1);
    float* den1  = (float*)(ws + off_den1);
    float* agg1  = (float*)(ws + off_agg1);

    const int B = 256;
    const float NEG_INF = -__builtin_huge_valf();

    // ======================= layer 0 ========================================
    cvt_f16_kernel<<<gridFor(N * FIN, B), B, 0, stream>>>(x, xh, N * FIN);
    cvt_w_t_kernel<<<gridFor((long long)FIN * HID, B), B, 0, stream>>>(W0, W0t, FIN, HID);
    cvt_w_t_kernel<<<gridFor((long long)HID * C1, B), B, 0, stream>>>(W1, W1t, HID, C1);

    {   // h0 = x @ W0   (wave-per-tile WMMA GEMM)
        long long waves = ((N + 15) / 16) * (HID / 16);
        wmma_gemm_kernel<<<gridFor(waves, 4), 128, 0, stream>>>(
            xh, W0t, feat0, (int)N, HID, FIN);
    }
    alpha_kernel<<<gridFor(N * H0, B), B, 0, stream>>>(feat0, attS0, attD0,
                                                       as0, ad0, N, H0, D0);
    fill_kernel<<<gridFor(N * H0, B), B, 0, stream>>>(m0, NEG_INF, N * H0);
    fill_kernel<<<gridFor(N * H0, B), B, 0, stream>>>(den0, 0.f, N * H0);
    fill_kernel<<<gridFor(N * HID, B), B, 0, stream>>>(agg0, 0.f, N * HID);

    edge_max_kernel<<<gridFor(E2 * H0, B), B, 0, stream>>>(
        src, dst, E, N, H0, as0, ad0, m0);
    edge_expsum_kernel<<<gridFor(E2 * H0, B), B, 0, stream>>>(
        src, dst, E, N, H0, as0, ad0, m0, wbuf, den0);
    edge_agg_kernel<<<gridFor(E2 * (H0 * D0), B), B, 0, stream>>>(
        src, dst, E, N, H0, D0, wbuf, den0, feat0, agg0);

    bias_elu_cvt_kernel<<<gridFor(N * HID, B), B, 0, stream>>>(agg0, b0, hh, N, HID);

    // ======================= layer 1 ========================================
    {   // h1 = hh @ W1
        long long waves = ((N + 15) / 16) * (C1 / 16);
        wmma_gemm_kernel<<<gridFor(waves, 4), 128, 0, stream>>>(
            hh, W1t, feat1, (int)N, C1, HID);
    }
    alpha_kernel<<<gridFor(N, B), B, 0, stream>>>(feat1, attS1, attD1,
                                                  as1, ad1, N, 1, C1);
    fill_kernel<<<gridFor(N, B), B, 0, stream>>>(m1, NEG_INF, N);
    fill_kernel<<<gridFor(N, B), B, 0, stream>>>(den1, 0.f, N);
    fill_kernel<<<gridFor(N * C1, B), B, 0, stream>>>(agg1, 0.f, N * C1);

    edge_max_kernel<<<gridFor(E2, B), B, 0, stream>>>(
        src, dst, E, N, 1, as1, ad1, m1);
    edge_expsum_kernel<<<gridFor(E2, B), B, 0, stream>>>(
        src, dst, E, N, 1, as1, ad1, m1, wbuf, den1);
    edge_agg_kernel<<<gridFor(E2 * C1, B), B, 0, stream>>>(
        src, dst, E, N, 1, C1, wbuf, den1, feat1, agg1);

    add_bias_kernel<<<gridFor(N * C1, B), B, 0, stream>>>(agg1, b1, out, N, C1);

    (void)n_in; (void)out_size; (void)ws_size;
}
